// MLPDecoder_584115552796
// MI455X (gfx1250) — compile-verified, matrix operand order checked by reading, not run on hardware
//
#include <hip/hip_runtime.h>

typedef __attribute__((ext_vector_type(16))) _Float16 v16h;
typedef __attribute__((ext_vector_type(8)))  _Float16 v8h;
typedef __attribute__((ext_vector_type(8)))  float    v8f;

#define N_TOK 1024
#define D_IN  256
#define H_DIM 256

// ---------------------------------------------------------------------------
// Kernel 1: f32 -> f16 conversion (z and W1 into workspace)
// ---------------------------------------------------------------------------
__global__ void cvt_f32_f16(const float* __restrict__ in, _Float16* __restrict__ out, int n) {
  int i = blockIdx.x * blockDim.x + threadIdx.x;
  if (i < n) out[i] = (_Float16)in[i];
}

// ---------------------------------------------------------------------------
// Kernel 2: fused front GEMM via v_wmma_f32_16x16x32_f16.
//   P[:, 0:256]   = z @ W1[:, :D]^T + b1   (row part, bias folded)
//   P[:, 256:512] = z @ W1[:, D:]^T        (col part)
// One wave (32 threads) computes one 16x16 output tile, K-loop of 8 WMMAs.
// A fragment: lane(0..15)=row m, holds K 0..7 (v0..3) + 16..23 (v4..7);
// lanes 16..31 same rows, K +8.  B fragment mirrors with lane=column n,
// and column n of B == row (h0+n) of W1 (contiguous along K).
// ---------------------------------------------------------------------------
__global__ void gemm16_wmma(const _Float16* __restrict__ A,   // [1024 x 256] f16
                            const _Float16* __restrict__ W,   // [256 x 512]  f16
                            const float* __restrict__ b1,     // [256]
                            float* __restrict__ P) {          // [1024 x 512] f32
  const int i0   = blockIdx.x * 16;            // output row tile
  const int hp0  = blockIdx.y * 16;            // output col tile in [0,512)
  const int lane = threadIdx.x & 31;
  const int half = lane >> 4;                  // 0 / 1
  const int l    = lane & 15;

  const int h0   = hp0 & (H_DIM - 1);          // W1 row base for this tile
  const int dOff = (hp0 >= H_DIM) ? D_IN : 0;  // col-part reads W1[:, D:]

  const _Float16* aRow = A + (size_t)(i0 + l) * D_IN;
  const _Float16* bRow = W + (size_t)(h0 + l) * (2 * D_IN) + dOff;

  v8f acc = {};
  for (int k = 0; k < D_IN; k += 32) {
    const _Float16* ap = aRow + k + half * 8;
    const _Float16* bp = bRow + k + half * 8;
    v8h alo = *(const v8h*)(ap);
    v8h ahi = *(const v8h*)(ap + 16);
    v8h blo = *(const v8h*)(bp);
    v8h bhi = *(const v8h*)(bp + 16);
    v16h a, b;
#pragma unroll
    for (int e = 0; e < 8; ++e) {
      a[e] = alo[e]; a[e + 8] = ahi[e];
      b[e] = blo[e]; b[e + 8] = bhi[e];
    }
    // (neg_a, A, neg_b, B, c_mod, C, reuse_a, reuse_b)
    acc = __builtin_amdgcn_wmma_f32_16x16x32_f16(false, a, false, b,
                                                 (short)0, acc, false, false);
  }

  // C/D layout: lane 0..15 -> N=lane, M=v; lane 16..31 -> N=lane-16, M=v+8
  const float bias = dOff ? 0.0f : b1[h0 + l];
#pragma unroll
  for (int v = 0; v < 8; ++v) {
    const int m = v + 8 * half;
    P[(size_t)(i0 + m) * (2 * H_DIM) + hp0 + l] = acc[v] + bias;
  }
}

// ---------------------------------------------------------------------------
// Kernel 3: pairwise relu-dot-sigmoid (inherently VALU: relu blocks matmul
// factoring).  64x64 output tile per 256-thread block, 4x4 register tile per
// thread, H chunked by 64 through LDS stored h-major so the inner loop does
// two conflict-free float4 LDS reads + one broadcast per h.
// ---------------------------------------------------------------------------
#define TI 64
#define TJ 64
#define HC 64

__global__ __launch_bounds__(256) void pairwise_relu_dot(
    const float* __restrict__ P,    // [1024 x 512]  (Rb | C)
    const float* __restrict__ W2,   // [256]
    const float* __restrict__ b2,   // [1]
    float* __restrict__ out) {      // [1024 x 1024]
  __shared__ float sA[HC][TI];      // sA[h][i_local]
  __shared__ float sB[HC][TJ];      // sB[h][j_local]
  __shared__ float sw[HC];

  const int tid = threadIdx.x;
  const int tx  = tid & 15;         // j group
  const int ty  = tid >> 4;         // i group
  const int i0  = blockIdx.y * TI;
  const int j0  = blockIdx.x * TJ;

  float acc[4][4];
#pragma unroll
  for (int a = 0; a < 4; ++a)
#pragma unroll
    for (int c = 0; c < 4; ++c) acc[a][c] = 0.0f;

  const int lrow = tid & 63;        // tile row this thread stages
  const int hgrp = tid >> 6;        // 0..3, 16 h-values each

  for (int hc = 0; hc < H_DIM; hc += HC) {
    if (tid < HC) sw[tid] = W2[hc + tid];
#pragma unroll
    for (int q = 0; q < 4; ++q) {
      const int h = hgrp * 16 + q * 4;
      float4 va = *(const float4*)(P + (size_t)(i0 + lrow) * (2 * H_DIM) + hc + h);
      float4 vb = *(const float4*)(P + (size_t)(j0 + lrow) * (2 * H_DIM) + H_DIM + hc + h);
      sA[h + 0][lrow] = va.x; sA[h + 1][lrow] = va.y;
      sA[h + 2][lrow] = va.z; sA[h + 3][lrow] = va.w;
      sB[h + 0][lrow] = vb.x; sB[h + 1][lrow] = vb.y;
      sB[h + 2][lrow] = vb.z; sB[h + 3][lrow] = vb.w;
    }
    __syncthreads();

#pragma unroll 4
    for (int h = 0; h < HC; ++h) {
      float4 av = *(const float4*)(&sA[h][ty * 4]);   // broadcast across tx
      float4 cv = *(const float4*)(&sB[h][tx * 4]);   // conflict-free across tx
      const float wh = sw[h];
      float ar[4] = {av.x, av.y, av.z, av.w};
      float cr[4] = {cv.x, cv.y, cv.z, cv.w};
#pragma unroll
      for (int a = 0; a < 4; ++a)
#pragma unroll
        for (int c = 0; c < 4; ++c) {
          float t = fmaxf(ar[a] + cr[c], 0.0f);
          acc[a][c] = fmaf(t, wh, acc[a][c]);
        }
    }
    __syncthreads();
  }

  const float bias2 = b2[0];
#pragma unroll
  for (int a = 0; a < 4; ++a) {
    float r[4];
#pragma unroll
    for (int c = 0; c < 4; ++c) {
      const float x = acc[a][c] + bias2;
      r[c] = 1.0f / (1.0f + __expf(-x));
    }
    float4 o = make_float4(r[0], r[1], r[2], r[3]);
    *(float4*)(out + (size_t)(i0 + ty * 4 + a) * N_TOK + j0 + tx * 4) = o;
  }
}

// ---------------------------------------------------------------------------
// Launch
// ---------------------------------------------------------------------------
extern "C" void kernel_launch(void* const* d_in, const int* in_sizes, int n_in,
                              void* d_out, int out_size, void* d_ws, size_t ws_size,
                              hipStream_t stream) {
  const float* z  = (const float*)d_in[0];   // [1024, 256]
  const float* W1 = (const float*)d_in[1];   // [256, 512]
  const float* b1 = (const float*)d_in[2];   // [256]
  const float* W2 = (const float*)d_in[3];   // [1, 256]
  const float* b2 = (const float*)d_in[4];   // [1]
  float* out = (float*)d_out;                // [1024, 1024]

  char* ws = (char*)d_ws;
  _Float16* zh  = (_Float16*)ws;                                  // 512 KB
  _Float16* W1h = (_Float16*)(ws + 512 * 1024);                   // 256 KB
  float*    P   = (float*)(ws + 512 * 1024 + 256 * 1024);         // 2 MB

  const int nz = N_TOK * D_IN;        // 262144
  const int nw = H_DIM * 2 * D_IN;    // 131072
  cvt_f32_f16<<<(nz + 255) / 256, 256, 0, stream>>>(z, zh, nz);
  cvt_f32_f16<<<(nw + 255) / 256, 256, 0, stream>>>(W1, W1h, nw);

  gemm16_wmma<<<dim3(N_TOK / 16, (2 * H_DIM) / 16), 32, 0, stream>>>(zh, W1h, b1, P);

  pairwise_relu_dot<<<dim3(N_TOK / TJ, N_TOK / TI), 256, 0, stream>>>(P, W2, b2, out);
}
